// GraphAttentionLayer_25907242730084
// MI455X (gfx1250) — matrix operand order repaired and checked by reading, not compiled, not run on hardware
//
#include <hip/hip_runtime.h>

// ---------------------------------------------------------------------------
// GAT layer for MI455X (gfx1250, wave32).
// Roofline: adj_in+adj_out int32 reads = 512MB -> ~22us at 23.3TB/s HBM.
// Strategy: read adj ONCE (stats pass, compress to 8MB/branch bitmask),
// then flash-style att@h with f16 WMMA (f32 accumulate) reading only the
// L2-resident bitmask + f16 h^T. f16 WMMA keeps the 17 GFLOP of matmul well
// under the memory floor (f32 WMMA 16x16x4 would not).
// ---------------------------------------------------------------------------

typedef __attribute__((ext_vector_type(16))) _Float16 v16h;
typedef __attribute__((ext_vector_type(8)))  _Float16 v8h;
typedef __attribute__((ext_vector_type(8)))  float    v8f;
typedef __attribute__((ext_vector_type(4)))  float    v4f;

#define NN       8192
#define F_IN     256
#define F_OUT    64
#define ALPHA_LR 0.2f
#define NEG_BIG  -9.0e15f
#define JSPLIT   4
#define JCHUNK   (NN / JSPLIT)   // 2048

// ---------------------------------------------------------------------------
// Kernel A: h = x @ W.  M=8192,K=256,N=64.
// Block = 256 thr = 8 waves; wave owns a 16x64 tile (4 accumulators).
// W is converted to f16 and stored transposed [n][k] in LDS so B-fragments
// are contiguous-K ds_load_b128 reads. h stored f32 (scores) + f16^T (pass2).
// ---------------------------------------------------------------------------
__global__ __launch_bounds__(256) void gat_gemm_h(
    const float* __restrict__ x, const float* __restrict__ W,
    float* __restrict__ h32, _Float16* __restrict__ hT16) {
  __shared__ _Float16 sWT[F_OUT * F_IN];   // 32KB, [n][k]
  const int tid = threadIdx.x;
  for (int idx = tid; idx < F_IN * F_OUT; idx += 256) {
    int k = idx >> 6, n = idx & 63;
    sWT[n * F_IN + k] = (_Float16)W[idx];
  }
  __syncthreads();

  const int wave = tid >> 5, lane = tid & 31;
  const int half = lane >> 4, l16 = lane & 15;
  const int mbase = blockIdx.x * 128 + wave * 16;
  const float* xr = x + (size_t)(mbase + l16) * F_IN;

  v8f c[4] = {v8f{}, v8f{}, v8f{}, v8f{}};
  for (int kb = 0; kb < F_IN; kb += 32) {
    // A fragment (16x32 f16): lane<16 rows m, K = 0..7 & 16..23 (rel kb);
    // lane>=16 same rows, K = 8..15 & 24..31.
    int k0 = kb + half * 8;
    v4f xa = *(const v4f*)(xr + k0);
    v4f xb = *(const v4f*)(xr + k0 + 4);
    v4f xc = *(const v4f*)(xr + k0 + 16);
    v4f xd = *(const v4f*)(xr + k0 + 20);
    v16h a;
#pragma unroll
    for (int i = 0; i < 4; ++i) {
      a[i]      = (_Float16)xa[i];
      a[4 + i]  = (_Float16)xb[i];
      a[8 + i]  = (_Float16)xc[i];
      a[12 + i] = (_Float16)xd[i];
    }
    int krow = kb + half * 8;  // B: K = half*8+0..7 & 16+half*8+0..7 for column l16
#pragma unroll
    for (int n = 0; n < 4; ++n) {
      const _Float16* bp = &sWT[(n * 16 + l16) * F_IN + krow];
      v8h blo = *(const v8h*)bp;
      v8h bhi = *(const v8h*)(bp + 16);
      v16h b;
#pragma unroll
      for (int i = 0; i < 8; ++i) { b[i] = blo[i]; b[8 + i] = bhi[i]; }
      c[n] = __builtin_amdgcn_wmma_f32_16x16x32_f16(
          false, a, false, b, (short)0, c[n], false, false);
    }
  }
  // C layout: VGPR r -> row (r) lanes 0-15, row (8+r) lanes 16-31; col = l16.
#pragma unroll
  for (int r = 0; r < 8; ++r) {
    int ro = mbase + r + half * 8;
#pragma unroll
    for (int n = 0; n < 4; ++n) {
      float v = c[n][r];
      int col = n * 16 + l16;
      h32[(size_t)ro * F_OUT + col] = v;
      hT16[(size_t)col * NN + ro] = (_Float16)v;
    }
  }
}

// ---------------------------------------------------------------------------
// Kernel B: ssrc[i] = h[i].a_src, sdst[i] = h[i].a_dst
// ---------------------------------------------------------------------------
__global__ __launch_bounds__(256) void gat_scores(
    const float* __restrict__ h32, const float* __restrict__ a,
    float* __restrict__ ssrc, float* __restrict__ sdst) {
  int i = blockIdx.x * 256 + threadIdx.x;
  const float* hr = h32 + (size_t)i * F_OUT;
  float s0 = 0.f, s1 = 0.f;
#pragma unroll
  for (int n = 0; n < F_OUT; ++n) {
    float hv = hr[n];
    s0 += hv * a[n];
    s1 += hv * a[F_OUT + n];
  }
  ssrc[i] = s0;
  sdst[i] = s1;
}

// ---------------------------------------------------------------------------
// Kernel C: single pass over adj (the 512MB). One wave per row.
// int4 loads (512B/wave-instr), compress adj>0 to bitmask (bit j%32 of word
// j/32), online (rowmax, rowsumexp) with exact NEG_BIG semantics of the
// reference (fully-masked rows degrade to uniform softmax, like jnp.where).
// ---------------------------------------------------------------------------
__global__ __launch_bounds__(256) void gat_stats(
    const int* __restrict__ adj, const float* __restrict__ ssrc,
    const float* __restrict__ sdst, unsigned* __restrict__ mask,
    float* __restrict__ rowmax, float* __restrict__ rowsum) {
  const int wave = threadIdx.x >> 5, lane = threadIdx.x & 31;
  const int i = blockIdx.x * 8 + wave;
  const float si = ssrc[i];
  const int* ar = adj + (size_t)i * NN;
  unsigned* mr = mask + (size_t)i * (NN / 32);

  float m = -3.0e38f, s = 0.f;
  for (int t = 0; t < NN / 128; ++t) {
    int j = t * 128 + lane * 4;
    int4 ad = *(const int4*)(ar + j);
    v4f sd = *(const v4f*)(sdst + j);
    // bitmask: lane contributes nibble at bits (lane%8)*4 of word lane/8
    unsigned nib = (ad.x > 0 ? 1u : 0u) | (ad.y > 0 ? 2u : 0u) |
                   (ad.z > 0 ? 4u : 0u) | (ad.w > 0 ? 8u : 0u);
    unsigned word = nib << ((lane & 7) * 4);
    word |= __shfl_xor(word, 1, 32);
    word |= __shfl_xor(word, 2, 32);
    word |= __shfl_xor(word, 4, 32);
    if ((lane & 7) == 0) mr[t * 4 + (lane >> 3)] = word;
    // online softmax stats (branchless lrelu: max(x, 0.2x))
    float e0 = si + sd[0]; e0 = fmaxf(e0, ALPHA_LR * e0);
    float e1 = si + sd[1]; e1 = fmaxf(e1, ALPHA_LR * e1);
    float e2 = si + sd[2]; e2 = fmaxf(e2, ALPHA_LR * e2);
    float e3 = si + sd[3]; e3 = fmaxf(e3, ALPHA_LR * e3);
    float v0 = (ad.x > 0) ? e0 : NEG_BIG;
    float v1 = (ad.y > 0) ? e1 : NEG_BIG;
    float v2 = (ad.z > 0) ? e2 : NEG_BIG;
    float v3 = (ad.w > 0) ? e3 : NEG_BIG;
    float mv = fmaxf(fmaxf(v0, v1), fmaxf(v2, v3));
    float mn = fmaxf(m, mv);
    s = s * __expf(m - mn) + __expf(v0 - mn) + __expf(v1 - mn) +
        __expf(v2 - mn) + __expf(v3 - mn);
    m = mn;
  }
  // wave butterfly merge
#pragma unroll
  for (int off = 16; off > 0; off >>= 1) {
    float om = __shfl_xor(m, off, 32);
    float os = __shfl_xor(s, off, 32);
    float mn = fmaxf(m, om);
    s = s * __expf(m - mn) + os * __expf(om - mn);
    m = mn;
  }
  if (lane == 0) { rowmax[i] = m; rowsum[i] = s; }
}

// ---------------------------------------------------------------------------
// Kernel D: out_partial = att @ h over a j-slice. Flash-style: global row
// max/sum already known, so j-slices just add (deterministic partials).
// Block = 8 waves x 16 rows; per 32-j step: h tile (32x64 f16) staged in LDS
// from hT16; P fragment built in registers in exact WMMA A-layout.
// ---------------------------------------------------------------------------
__global__ __launch_bounds__(256) void gat_flash(
    const _Float16* __restrict__ hT16, const float* __restrict__ ssrc,
    const float* __restrict__ sdst_g, const unsigned* __restrict__ mask,
    const float* __restrict__ rowmax, const float* __restrict__ rowsum,
    float* __restrict__ obp) {
  __shared__ float sdst_s[JCHUNK];          // 8KB slice
  __shared__ _Float16 htile[F_OUT * 32];    // 4KB, [n][jlocal]
  const int tid = threadIdx.x;
  const int jbase = blockIdx.y * JCHUNK;
  for (int idx = tid; idx < JCHUNK; idx += 256) sdst_s[idx] = sdst_g[jbase + idx];

  const int wave = tid >> 5, lane = tid & 31;
  const int half = lane >> 4, l16 = lane & 15;
  const int mtile = blockIdx.x * 128 + wave * 16;
  const int mrow = mtile + l16;
  const float sm = ssrc[mrow];
  const float rmx = rowmax[mrow];
  const unsigned* mr = mask + (size_t)mrow * (NN / 32);
  const int ln = tid >> 2;        // 0..63: h column this thread stages
  const int lc = (tid & 3) * 8;   // j-sub-chunk
  const int off = half * 8;

  v8f c[4] = {v8f{}, v8f{}, v8f{}, v8f{}};
  __syncthreads();
  for (int jl = 0; jl < JCHUNK; jl += 32) {
    v8h hv = *(const v8h*)(hT16 + (size_t)ln * NN + jbase + jl + lc);
    __syncthreads();                       // previous tile fully consumed
    *(v8h*)(htile + ln * 32 + lc) = hv;
    __syncthreads();                       // tile visible

    unsigned word = mr[(jbase + jl) >> 5];
    v16h a;
    const float* sd0 = &sdst_s[jl + off];
    const float* sd1 = &sdst_s[jl + 16 + off];
#pragma unroll
    for (int q = 0; q < 8; ++q) {          // A elems 0..7: K = half*8 + q
      float e = sm + sd0[q];
      e = fmaxf(e, ALPHA_LR * e);
      float v = ((word >> (off + q)) & 1u) ? e : NEG_BIG;
      a[q] = (_Float16)__expf(v - rmx);
    }
#pragma unroll
    for (int q = 0; q < 8; ++q) {          // A elems 8..15: K = 16+half*8+q
      float e = sm + sd1[q];
      e = fmaxf(e, ALPHA_LR * e);
      float v = ((word >> (16 + off + q)) & 1u) ? e : NEG_BIG;
      a[8 + q] = (_Float16)__expf(v - rmx);
    }
#pragma unroll
    for (int n = 0; n < 4; ++n) {
      const _Float16* bp = &htile[(n * 16 + l16) * 32 + half * 8];
      v8h blo = *(const v8h*)bp;
      v8h bhi = *(const v8h*)(bp + 16);
      v16h b;
#pragma unroll
      for (int i = 0; i < 8; ++i) { b[i] = blo[i]; b[8 + i] = bhi[i]; }
      c[n] = __builtin_amdgcn_wmma_f32_16x16x32_f16(
          false, a, false, b, (short)0, c[n], false, false);
    }
  }
  float* op = obp + (size_t)blockIdx.y * (NN * F_OUT);
#pragma unroll
  for (int r = 0; r < 8; ++r) {
    int ro = mtile + r + half * 8;
    float inv = 1.0f / rowsum[ro];
#pragma unroll
    for (int n = 0; n < 4; ++n)
      op[(size_t)ro * F_OUT + n * 16 + l16] = c[n][r] * inv;
  }
}

// ---------------------------------------------------------------------------
// Kernel E: sum j-partials, average branches, ELU.
// ---------------------------------------------------------------------------
__global__ __launch_bounds__(256) void gat_combine(
    const float* __restrict__ op_in, const float* __restrict__ op_out,
    float* __restrict__ out) {
  int i = blockIdx.x * 256 + threadIdx.x;
  float s = 0.f;
#pragma unroll
  for (int p = 0; p < JSPLIT; ++p)
    s += op_in[(size_t)p * (NN * F_OUT) + i] + op_out[(size_t)p * (NN * F_OUT) + i];
  float v = 0.5f * s;
  out[i] = v > 0.f ? v : (__expf(v) - 1.f);
}

// ---------------------------------------------------------------------------
extern "C" void kernel_launch(void* const* d_in, const int* in_sizes, int n_in,
                              void* d_out, int out_size, void* d_ws, size_t ws_size,
                              hipStream_t stream) {
  const float* x       = (const float*)d_in[0];
  const int*   adj_in  = (const int*)d_in[1];
  const int*   adj_out = (const int*)d_in[2];
  const float* W_in    = (const float*)d_in[3];
  const float* W_out   = (const float*)d_in[4];
  const float* a       = (const float*)d_in[5];

  char* ws = (char*)d_ws;
  size_t off = 0;
  auto alloc = [&](size_t bytes) -> char* {
    char* p = ws + off;
    off += (bytes + 255) & ~(size_t)255;
    return p;
  };
  float*    h32_in   = (float*)alloc((size_t)NN * F_OUT * 4);
  float*    h32_out  = (float*)alloc((size_t)NN * F_OUT * 4);
  _Float16* hT16_in  = (_Float16*)alloc((size_t)NN * F_OUT * 2);
  _Float16* hT16_out = (_Float16*)alloc((size_t)NN * F_OUT * 2);
  float*    ssrc_in  = (float*)alloc(NN * 4);
  float*    sdst_in  = (float*)alloc(NN * 4);
  float*    ssrc_out = (float*)alloc(NN * 4);
  float*    sdst_out = (float*)alloc(NN * 4);
  float*    rmax_in  = (float*)alloc(NN * 4);
  float*    rsum_in  = (float*)alloc(NN * 4);
  float*    rmax_out = (float*)alloc(NN * 4);
  float*    rsum_out = (float*)alloc(NN * 4);
  unsigned* mask_in  = (unsigned*)alloc((size_t)NN * (NN / 32) * 4);
  unsigned* mask_out = (unsigned*)alloc((size_t)NN * (NN / 32) * 4);
  float*    op_in    = (float*)alloc((size_t)JSPLIT * NN * F_OUT * 4);
  float*    op_out   = (float*)alloc((size_t)JSPLIT * NN * F_OUT * 4);
  (void)ws_size; (void)in_sizes; (void)n_in; (void)out_size;

  gat_gemm_h<<<NN / 128, 256, 0, stream>>>(x, W_in, h32_in, hT16_in);
  gat_gemm_h<<<NN / 128, 256, 0, stream>>>(x, W_out, h32_out, hT16_out);

  gat_scores<<<NN / 256, 256, 0, stream>>>(h32_in, a, ssrc_in, sdst_in);
  gat_scores<<<NN / 256, 256, 0, stream>>>(h32_out, a, ssrc_out, sdst_out);

  gat_stats<<<NN / 8, 256, 0, stream>>>(adj_in, ssrc_in, sdst_in,
                                        mask_in, rmax_in, rsum_in);
  gat_stats<<<NN / 8, 256, 0, stream>>>(adj_out, ssrc_out, sdst_out,
                                        mask_out, rmax_out, rsum_out);

  dim3 fgrid(NN / 128, JSPLIT);
  gat_flash<<<fgrid, 256, 0, stream>>>(hT16_in, ssrc_in, sdst_in,
                                       mask_in, rmax_in, rsum_in, op_in);
  gat_flash<<<fgrid, 256, 0, stream>>>(hT16_out, ssrc_out, sdst_out,
                                       mask_out, rmax_out, rsum_out, op_out);

  gat_combine<<<(NN * F_OUT) / 256, 256, 0, stream>>>(op_in, op_out, (float*)d_out);
}
